// _ExactIntegerLinear_6631429505166
// MI455X (gfx1250) — 8-bit matrix operand order repaired and checked by reading, compiled, not run on hardware
//
#include <hip/hip_runtime.h>
#include <cstdint>
#include <cstddef>

typedef __attribute__((ext_vector_type(8))) int      v8i;
typedef __attribute__((ext_vector_type(4))) int      v4i;
typedef __attribute__((ext_vector_type(4))) unsigned v4u;
typedef __attribute__((ext_vector_type(4))) float    v4f;

#define IN_F   4096
#define OUT_F  4096
#define M_ROWS 16384

#if defined(__has_builtin)
#  if __has_builtin(__builtin_amdgcn_global_load_async_to_lds_b128)
#    define HAVE_ASYNC_LDS 1
#  endif
#endif
#ifndef HAVE_ASYNC_LDS
#  define HAVE_ASYNC_LDS 0
#endif

__device__ __forceinline__ void async_copy_b128(const void* g, void* l) {
#if HAVE_ASYNC_LDS
  __builtin_amdgcn_global_load_async_to_lds_b128(
      (__attribute__((address_space(1))) v4i*)(g),
      (__attribute__((address_space(3))) v4i*)(l), 0, 0);
#else
  *(v4u*)l = *(const v4u*)g;
#endif
}

__device__ __forceinline__ void async_wait_all() {
#if HAVE_ASYNC_LDS
#  if __has_builtin(__builtin_amdgcn_s_wait_asynccnt)
  __builtin_amdgcn_s_wait_asynccnt(0);
#  else
  asm volatile("s_wait_asynccnt 0" ::: "memory");
#  endif
#endif
}

__global__ void zero_i32(int* __restrict__ p, int n) {
  int i = blockIdx.x * blockDim.x + threadIdx.x;
  if (i < n) p[i] = 0;
}

__global__ void quant_pack_x(const float* __restrict__ x,
                             const float* __restrict__ plog_s,
                             const float* __restrict__ plog_q,
                             const float* __restrict__ pact_b,
                             unsigned char* __restrict__ xq_pack,
                             int* __restrict__ Sn)
{
  const float s     = exp2f(*plog_s);
  const float q     = exp2f(*plog_q);
  const float zp    = rintf(*pact_b / s * 2.0f) * 0.5f * s;
  const float lo    = zp;
  const float hi    = zp + q - s;
  const float inv_s = 1.0f / s;

  int tid  = blockIdx.x * blockDim.x + threadIdx.x;
  int lane = tid & 31;
  int tile = tid >> 5;
  int mt   = tile >> 6;
  int kt   = tile & 63;
  int m    = mt * 16 + (lane & 15);
  int h    = lane >> 4;

  unsigned dw[8];
  int rsum = 0;
#pragma unroll
  for (int j = 0; j < 8; ++j) {
    int kd = 4 * (j >> 1) + (j & 1) + 2 * h;
    const v4f xv = __builtin_nontemporal_load(
        (const v4f*)(x + (size_t)m * IN_F + (size_t)(kt * 16 + kd) * 4));
    unsigned d = 0;
#pragma unroll
    for (int b = 0; b < 4; ++b) {
      float v = xv[b];
      v = fminf(fmaxf(v, lo), hi);
      int code = (int)rintf((v - zp) * inv_s);
      rsum += code;
      d |= ((unsigned)code & 0xffu) << (8 * b);
    }
    dw[j] = d;
  }
  size_t base = (size_t)tile * 1024 + (size_t)lane * 32;
  *(v4u*)(xq_pack + base)      = (v4u){dw[0], dw[1], dw[2], dw[3]};
  *(v4u*)(xq_pack + base + 16) = (v4u){dw[4], dw[5], dw[6], dw[7]};
  atomicAdd(&Sn[m], rsum);
}

__global__ void pack_w(const float* __restrict__ w,
                       unsigned char* __restrict__ w_pack,
                       int* __restrict__ Sw)
{
  int tid  = blockIdx.x * blockDim.x + threadIdx.x;
  int lane = tid & 31;
  int tile = tid >> 5;
  int nt   = tile >> 6;
  int kt   = tile & 63;
  int o    = nt * 16 + (lane & 15);
  int h    = lane >> 4;

  unsigned dw[8];
  int rsum = 0;
#pragma unroll
  for (int j = 0; j < 8; ++j) {
    int kd = 4 * (j >> 1) + (j & 1) + 2 * h;
    const v4f wv = __builtin_nontemporal_load(
        (const v4f*)(w + (size_t)o * IN_F + (size_t)(kt * 16 + kd) * 4));
    unsigned d = 0;
#pragma unroll
    for (int b = 0; b < 4; ++b) {
      int wi = (int)rintf(wv[b]);
      wi = (wi < -128) ? -128 : ((wi > 127) ? 127 : wi);
      rsum += wi;
      d |= ((unsigned)wi & 0xffu) << (8 * b);
    }
    dw[j] = d;
  }
  size_t base = (size_t)tile * 1024 + (size_t)lane * 32;
  *(v4u*)(w_pack + base)      = (v4u){dw[0], dw[1], dw[2], dw[3]};
  *(v4u*)(w_pack + base + 16) = (v4u){dw[4], dw[5], dw[6], dw[7]};
  atomicAdd(&Sw[o], rsum);
}

__global__ void make_consts(const float* __restrict__ bias,
                            const float* __restrict__ scale,
                            const float* __restrict__ zpw,
                            const int*   __restrict__ Sw,
                            const float* __restrict__ plog_s,
                            const float* __restrict__ pact_b,
                            float* __restrict__ c0,
                            float* __restrict__ c1,
                            float* __restrict__ c2)
{
  int o = blockIdx.x * blockDim.x + threadIdx.x;
  if (o >= OUT_F) return;
  float s  = exp2f(*plog_s);
  float zp = rintf(*pact_b / s * 2.0f) * 0.5f * s;
  float sc = scale[o];
  float z  = zpw[o];
  c1[o] = sc * s;
  c2[o] = z * s;
  c0[o] = sc * zp * (float)Sw[o] + z * zp * (float)IN_F + bias[o];
}

__global__ __launch_bounds__(256) void gemm_iu8(
    const unsigned char* __restrict__ xq_pack,
    const unsigned char* __restrict__ w_pack,
    const int*   __restrict__ Sn,
    const float* __restrict__ c0,
    const float* __restrict__ c1,
    const float* __restrict__ c2,
    float* __restrict__ out)
{
  __shared__ alignas(16) unsigned char sB[2][4096];

  const int tid  = threadIdx.x;
  const int lane = tid & 31;
  const int wv   = tid >> 5;
  const int bm   = blockIdx.x >> 6;
  const int bn   = blockIdx.x & 63;
  const int mt   = bm * 8 + wv;

  const int bt   = tid >> 6;
  const int boff = (tid & 63) * 16;
  const unsigned char* bsrc = w_pack + (size_t)(bn * 4 + bt) * (64 * 1024) + boff;
  unsigned char* bdst0 = &sB[0][bt * 1024 + boff];
  unsigned char* bdst1 = &sB[1][bt * 1024 + boff];

  async_copy_b128(bsrc, bdst0);

  const unsigned char* ap = xq_pack + (size_t)mt * (64 * 1024) + (size_t)lane * 32;

  v8i zero = {0, 0, 0, 0, 0, 0, 0, 0};
  v8i acc[4] = {zero, zero, zero, zero};

  for (int kt = 0; kt < 64; ++kt) {
    const int buf = kt & 1;
    async_wait_all();
    __syncthreads();
    if (kt + 1 < 64)
      async_copy_b128(bsrc + (size_t)(kt + 1) * 1024, buf ? bdst0 : bdst1);

    union { v4u u[2]; v8i v; } A;
    A.u[0] = *(const v4u*)(ap + (size_t)kt * 1024);
    A.u[1] = *(const v4u*)(ap + (size_t)kt * 1024 + 16);
    if (kt + 1 < 64)
      __builtin_prefetch(ap + (size_t)(kt + 1) * 1024, 0, 0);

#pragma unroll
    for (int t = 0; t < 4; ++t) {
      union { v4u u[2]; v8i v; } B;
      const v4u* bs = (const v4u*)(&sB[buf][t * 1024 + lane * 32]);
      B.u[0] = bs[0];
      B.u[1] = bs[1];
      acc[t] = __builtin_amdgcn_wmma_i32_16x16x64_iu8(
          false, A.v, true, B.v, acc[t], false, false);
    }
  }

  const int mrow0 = mt * 16 + ((lane >> 4) << 3);
  const int nb    = bn * 64 + (lane & 15);

  float snf[8];
#pragma unroll
  for (int r = 0; r < 8; ++r) snf[r] = (float)Sn[mrow0 + r];

#pragma unroll
  for (int t = 0; t < 4; ++t) {
    int n = nb + t * 16;
    float C0 = c0[n], C1 = c1[n], C2 = c2[n];
#pragma unroll
    for (int r = 0; r < 8; ++r) {
      int m = mrow0 + r;
      float v = C1 * (float)acc[t][r] + C2 * snf[r] + C0;
      __builtin_nontemporal_store(v, &out[(size_t)m * OUT_F + n]);
    }
  }
}

extern "C" void kernel_launch(void* const* d_in, const int* in_sizes, int n_in,
                              void* d_out, int out_size, void* d_ws, size_t ws_size,
                              hipStream_t stream)
{
  const float* x      = (const float*)d_in[0];
  const float* wgt    = (const float*)d_in[1];
  const float* bias   = (const float*)d_in[2];
  const float* wsc    = (const float*)d_in[3];
  const float* wzp    = (const float*)d_in[4];
  const float* plog_s = (const float*)d_in[5];
  const float* plog_q = (const float*)d_in[6];
  const float* pact_b = (const float*)d_in[7];
  float* out = (float*)d_out;

  char* ws = (char*)d_ws;
  unsigned char* xq_pack = (unsigned char*)ws;
  unsigned char* w_pack  = (unsigned char*)(ws + ((size_t)64 << 20));
  int*   Sn = (int*)(ws + ((size_t)80 << 20));
  int*   Sw = Sn + M_ROWS;
  float* c0 = (float*)(Sw + OUT_F);
  float* c1 = c0 + OUT_F;
  float* c2 = c1 + OUT_F;

  zero_i32<<<(M_ROWS + OUT_F + 255) / 256, 256, 0, stream>>>(Sn, M_ROWS + OUT_F);

  quant_pack_x<<<8192, 256, 0, stream>>>(x, plog_s, plog_q, pact_b, xq_pack, Sn);

  pack_w<<<2048, 256, 0, stream>>>(wgt, w_pack, Sw);

  make_consts<<<16, 256, 0, stream>>>(bias, wsc, wzp, Sw, plog_s, pact_b, c0, c1, c2);

  gemm_iu8<<<8192, 256, 0, stream>>>(xq_pack, w_pack, Sn, c0, c1, c2, out);
}
